// GRPOMoE_78709570667260
// MI455X (gfx1250) — matrix operand order repaired and checked by reading, not compiled, not run on hardware
//
#include <hip/hip_runtime.h>
#include <math.h>

// ---------------------------------------------------------------------------
// Types for WMMA
// ---------------------------------------------------------------------------
typedef _Float16 v8h  __attribute__((ext_vector_type(8)));
typedef _Float16 v16h __attribute__((ext_vector_type(16)));
typedef float    v8f  __attribute__((ext_vector_type(8)));

#define D_MODEL 512
#define D_FF    1024
#define N_EXP   8
#define N_LAYERS 2
#define NTOK    2048            // B*S = 2*1024

__device__ __forceinline__ float gelu_exact(float v) {
    return 0.5f * v * (1.0f + erff(v * 0.70710678118654752f));
}
__device__ __forceinline__ float sigmoidf_(float v) {
    return 1.0f / (1.0f + expf(-v));
}

// ---------------------------------------------------------------------------
// Tiled WMMA GEMM: C[M,N] = A[M,K] @ B[K,N] + bias, optional exact GELU.
// BM=128 BN=128 BK=32, 256 threads (8 waves), each wave computes 32x64 via
// 2x4 v_wmma_f32_16x16x32_f16 tiles. Software pipeline: register prefetch of
// tile k+1 overlaps WMMA on LDS buffer k%2; one barrier per K-step.
// All A/B fragments are materialized before the MMA block so the 12
// ds_load_b128 issue together and s_wait_dscnt staggers behind the WMMAs.
// LDS rows padded to 40 halves (80B = 20 dwords) to rotate banks.
// Requires M%128==0, N%128==0, K%32==0.
// ---------------------------------------------------------------------------
template <int EPI>   // 0 = none, 1 = gelu
__global__ __launch_bounds__(256)
void gemm_wmma(const float* __restrict__ A, const float* __restrict__ B,
               const float* __restrict__ bias, float* __restrict__ C,
               int M, int N, int K) {
    constexpr int BM = 128, BN = 128, BK = 32, BKP = 40;  // BKP = padded row
    __shared__ _Float16 sA[2][BM * BKP] __attribute__((aligned(64)));  // [m][k]
    __shared__ _Float16 sB[2][BN * BKP] __attribute__((aligned(64)));  // [n][k]

    const int tid  = threadIdx.x;
    const int wave = tid >> 5;
    const int lane = tid & 31;
    const int wm   = (wave & 3) * 32;     // wave M offset (4 waves along M)
    const int wn   = (wave >> 2) * 64;    // wave N offset (2 waves along N)
    const int m0   = blockIdx.y * BM;
    const int n0   = blockIdx.x * BN;
    const int lh   = lane & 15;           // row/col within 16-wide subtile
    const int lsel = lane >> 4;           // K half-group this lane owns

    // A staging coords: 4 chunks of float4 (row ar, k-col ac)
    int ar[4], ac[4];
    #pragma unroll
    for (int it = 0; it < 4; ++it) {
        const int i = tid + it * 256;
        ar[it] = i >> 3;  ac[it] = (i & 7) * 4;          // A: [row 0..127][k 0..31]
    }
    // B staging coords: this thread owns column nB, k-range [kh, kh+16)
    const int nB = tid & 127;
    const int kh = (tid >> 7) * 16;

    float4 ra[4];
    float  rbv[16];
    auto load_tile = [&](int k0) {
        #pragma unroll
        for (int it = 0; it < 4; ++it)
            ra[it] = *(const float4*)(A + (size_t)(m0 + ar[it]) * K + k0 + ac[it]);
        #pragma unroll
        for (int j = 0; j < 16; ++j)                      // coalesced along n
            rbv[j] = B[(size_t)(k0 + kh + j) * N + n0 + nB];
    };
    auto stage_tile = [&](int buf) {
        #pragma unroll
        for (int it = 0; it < 4; ++it) {
            _Float16* p = &sA[buf][ar[it] * BKP + ac[it]];
            p[0] = (_Float16)ra[it].x; p[1] = (_Float16)ra[it].y;
            p[2] = (_Float16)ra[it].z; p[3] = (_Float16)ra[it].w;
        }
        v8h b0, b1;
        #pragma unroll
        for (int j = 0; j < 8; ++j) {
            b0[j] = (_Float16)rbv[j];
            b1[j] = (_Float16)rbv[8 + j];
        }
        *(v8h*)&sB[buf][nB * BKP + kh]     = b0;          // merged -> ds_store_b128
        *(v8h*)&sB[buf][nB * BKP + kh + 8] = b1;
    };

    v8f acc[2][4];
    #pragma unroll
    for (int ms = 0; ms < 2; ++ms)
        #pragma unroll
        for (int ns = 0; ns < 4; ++ns) acc[ms][ns] = (v8f)(0.0f);

    const int nk = K / BK;
    load_tile(0);
    stage_tile(0);
    __syncthreads();

    for (int kt = 0; kt < nk; ++kt) {
        const int cur = kt & 1;
        if (kt + 1 < nk) load_tile((kt + 1) * BK);   // issue, don't wait

        // ---- A fragments (ISA 16-bit A 16x32 layout) ----
        v16h af[2];
        #pragma unroll
        for (int ms = 0; ms < 2; ++ms) {
            const _Float16* pa = &sA[cur][(wm + ms * 16 + lh) * BKP];
            const v8h lo = *(const v8h*)(pa + lsel * 8);        // K {0..7} / {8..15}
            const v8h hi = *(const v8h*)(pa + 16 + lsel * 8);   // K {16..23} / {24..31}
            af[ms] = __builtin_shufflevector(lo, hi, 0,1,2,3,4,5,6,7,
                                                      8,9,10,11,12,13,14,15);
        }
        // ---- all B fragments first (lanes 0-15: K 0..7,16..23, lanes 16-31: K 8..15,24..31) ----
        v16h bf[4];
        #pragma unroll
        for (int ns = 0; ns < 4; ++ns) {
            const _Float16* pb = &sB[cur][(wn + ns * 16 + lh) * BKP];
            const v8h blo = *(const v8h*)(pb + lsel * 8);
            const v8h bhi = *(const v8h*)(pb + 16 + lsel * 8);
            bf[ns] = __builtin_shufflevector(blo, bhi, 0,1,2,3,4,5,6,7,
                                                       8,9,10,11,12,13,14,15);
        }
        // ---- 8 WMMAs, loads already in flight ----
        #pragma unroll
        for (int ns = 0; ns < 4; ++ns) {
            #pragma unroll
            for (int ms = 0; ms < 2; ++ms) {
                acc[ms][ns] = __builtin_amdgcn_wmma_f32_16x16x32_f16(
                    false, af[ms], false, bf[ns], (short)0, acc[ms][ns], false, false);
            }
        }
        if (kt + 1 < nk) stage_tile((kt + 1) & 1);   // load-wait lands here, after WMMAs
        __syncthreads();
    }

    // ---- epilogue: bias (+gelu), store (C/D layout: VGPR r -> M = r or r+8) ----
    const int nc = lane & 15;
    const int mb = (lane >> 4) * 8;
    #pragma unroll
    for (int ms = 0; ms < 2; ++ms) {
        #pragma unroll
        for (int ns = 0; ns < 4; ++ns) {
            const int row0 = m0 + wm + ms * 16 + mb;
            const int col  = n0 + wn + ns * 16 + nc;
            const float bv = bias[col];
            float* pc = C + (size_t)row0 * N + col;
            const v8f a = acc[ms][ns];
            #pragma unroll
            for (int r = 0; r < 8; ++r) {
                float v = a[r] + bv;
                if (EPI == 1) v = gelu_exact(v);
                pc[(size_t)r * N] = v;
            }
        }
    }
}

// ---------------------------------------------------------------------------
// block-wide sum over 256 threads (wave32 shuffle + one LDS step)
// ---------------------------------------------------------------------------
__device__ __forceinline__ float block_sum256(float v, float* red, int tid) {
    #pragma unroll
    for (int off = 16; off > 0; off >>= 1) v += __shfl_xor(v, off, 32);
    if ((tid & 31) == 0) red[tid >> 5] = v;
    __syncthreads();
    float tot = 0.f;
    #pragma unroll
    for (int i = 0; i < 8; ++i) tot += red[i];
    __syncthreads();
    return tot;
}

// ---------------------------------------------------------------------------
// zero output
// ---------------------------------------------------------------------------
__global__ void zero_kernel(float* p, int n) {
    int i = blockIdx.x * blockDim.x + threadIdx.x;
    if (i < n) p[i] = 0.0f;
}

// ---------------------------------------------------------------------------
// Policy head LN + GELU (in place on Hp, rows of 256)
// ---------------------------------------------------------------------------
__global__ __launch_bounds__(256)
void policy_ln_gelu(float* __restrict__ Hp,
                    const float* __restrict__ g, const float* __restrict__ b) {
    const int t = blockIdx.x, tid = threadIdx.x;
    __shared__ float red[8];
    float v = Hp[(size_t)t * 256 + tid];
    const float mean = block_sum256(v, red, tid) * (1.0f / 256.0f);
    const float d = v - mean;
    const float var = block_sum256(d * d, red, tid) * (1.0f / 256.0f);
    const float rstd = rsqrtf(var + 1e-5f);
    const float y = d * rstd * g[tid] + b[tid];
    Hp[(size_t)t * 256 + tid] = gelu_exact(y);
}

// ---------------------------------------------------------------------------
// Routing: logits = Hp @ pw2 + pb2 ; gumbel (threefry2x32, key(42)) ; top-2
// ---------------------------------------------------------------------------
__device__ __forceinline__ unsigned rotl32(unsigned v, int r) {
    return (v << r) | (v >> (32 - r));
}
__device__ void threefry2x32(unsigned k0, unsigned k1,
                             unsigned& x0, unsigned& x1) {
    const unsigned ks0 = k0, ks1 = k1, ks2 = k0 ^ k1 ^ 0x1BD11BDAu;
    const int RA[4] = {13, 15, 26, 6};
    const int RB[4] = {17, 29, 16, 24};
    x0 += ks0; x1 += ks1;
    #pragma unroll
    for (int i = 0; i < 4; ++i) { x0 += x1; x1 = rotl32(x1, RA[i]); x1 ^= x0; }
    x0 += ks1; x1 += ks2 + 1u;
    #pragma unroll
    for (int i = 0; i < 4; ++i) { x0 += x1; x1 = rotl32(x1, RB[i]); x1 ^= x0; }
    x0 += ks2; x1 += ks0 + 2u;
    #pragma unroll
    for (int i = 0; i < 4; ++i) { x0 += x1; x1 = rotl32(x1, RA[i]); x1 ^= x0; }
    x0 += ks0; x1 += ks1 + 3u;
    #pragma unroll
    for (int i = 0; i < 4; ++i) { x0 += x1; x1 = rotl32(x1, RB[i]); x1 ^= x0; }
    x0 += ks1; x1 += ks2 + 4u;
    #pragma unroll
    for (int i = 0; i < 4; ++i) { x0 += x1; x1 = rotl32(x1, RA[i]); x1 ^= x0; }
    x0 += ks2; x1 += ks0 + 5u;
}

__device__ float jax_uniform_bits(int flat_idx) {
    // jax.random.uniform(key(42), (2,1024,8)): counts = iota(16384), split halves
    const unsigned half = 8192u;
    unsigned x0, x1;
    if ((unsigned)flat_idx < half) { x0 = (unsigned)flat_idx; x1 = (unsigned)flat_idx + half; }
    else                          { x0 = (unsigned)flat_idx - half; x1 = (unsigned)flat_idx; }
    threefry2x32(0u, 42u, x0, x1);
    unsigned bits = ((unsigned)flat_idx < half) ? x0 : x1;
    unsigned fb = (bits >> 9) | 0x3F800000u;
    return __uint_as_float(fb) - 1.0f;
}

__global__ __launch_bounds__(256)
void policy_route(const float* __restrict__ Hp, const float* __restrict__ pw2,
                  const float* __restrict__ pb2, float* __restrict__ sel) {
    const int t = blockIdx.x, tid = threadIdx.x;
    __shared__ float row[256];
    __shared__ float part[256];
    __shared__ float score[N_EXP];
    row[tid] = Hp[(size_t)t * 256 + tid];
    __syncthreads();
    const int e = tid & 7, j = tid >> 3;
    float p = 0.0f;
    #pragma unroll
    for (int q = 0; q < 8; ++q) {
        const int i = j * 8 + q;
        p += row[i] * pw2[(size_t)i * N_EXP + e];
    }
    part[tid] = p;
    __syncthreads();
    if (tid < N_EXP) {
        float logit = pb2[tid];
        for (int jj = 0; jj < 32; ++jj) logit += part[jj * 8 + tid];
        const float u = jax_uniform_bits(t * N_EXP + tid);
        const float g = -logf(-logf(u + 1e-10f) + 1e-10f);
        score[tid] = logit + g;   // TEMP = 1.0
    }
    __syncthreads();
    if (tid == 0) {
        int i1 = 0; float b1 = score[0];
        for (int k = 1; k < N_EXP; ++k) if (score[k] > b1) { b1 = score[k]; i1 = k; }
        int i2 = -1; float b2 = -1e30f;
        for (int k = 0; k < N_EXP; ++k)
            if (k != i1 && score[k] > b2) { b2 = score[k]; i2 = k; }
        for (int k = 0; k < N_EXP; ++k)
            sel[(size_t)t * N_EXP + k] = (k == i1 || k == i2) ? 1.0f : 0.0f;
    }
}

// ---------------------------------------------------------------------------
// Per-layer: gate = sigmoid(h . gw) ; h2 = LN(u) ; h += gate * h2
// ---------------------------------------------------------------------------
__global__ __launch_bounds__(256)
void gate_ln_residual(float* __restrict__ H, const float* __restrict__ U,
                      const float* __restrict__ ng, const float* __restrict__ nb,
                      const float* __restrict__ gw) {
    const int F = D_FF;
    const int t = blockIdx.x, tid = threadIdx.x;
    __shared__ float red[8];
    float h[4], u[4];
    float s1 = 0.f, s2 = 0.f, sg = 0.f;
    #pragma unroll
    for (int j = 0; j < 4; ++j) {
        const int i = tid + j * 256;
        h[j] = H[(size_t)t * F + i];
        u[j] = U[(size_t)t * F + i];
        s1 += u[j]; s2 += u[j] * u[j]; sg += h[j] * gw[i];
    }
    const float S1 = block_sum256(s1, red, tid);
    const float S2 = block_sum256(s2, red, tid);
    const float SG = block_sum256(sg, red, tid);
    const float mean = S1 / (float)F;
    const float var  = S2 / (float)F - mean * mean;
    const float rstd = rsqrtf(var + 1e-5f);
    const float gate = sigmoidf_(SG);
    #pragma unroll
    for (int j = 0; j < 4; ++j) {
        const int i = tid + j * 256;
        const float h2 = (u[j] - mean) * rstd * ng[i] + nb[i];
        H[(size_t)t * F + i] = h[j] + gate * h2;
    }
}

// ---------------------------------------------------------------------------
// Final: r = x + o ; eout = LN(r, fng, fnb) ; out += sel[t,e] * eout
// ---------------------------------------------------------------------------
__global__ __launch_bounds__(256)
void final_ln_accum(const float* __restrict__ X, const float* __restrict__ O,
                    const float* __restrict__ fng, const float* __restrict__ fnb,
                    const float* __restrict__ sel, int e, float* __restrict__ out) {
    const int D = D_MODEL;
    const int t = blockIdx.x, tid = threadIdx.x;
    __shared__ float red[8];
    float r[2];
    float s1 = 0.f, s2 = 0.f;
    #pragma unroll
    for (int j = 0; j < 2; ++j) {
        const int i = tid + j * 256;
        r[j] = X[(size_t)t * D + i] + O[(size_t)t * D + i];
        s1 += r[j]; s2 += r[j] * r[j];
    }
    const float S1 = block_sum256(s1, red, tid);
    const float S2 = block_sum256(s2, red, tid);
    const float mean = S1 / (float)D;
    const float var  = S2 / (float)D - mean * mean;
    const float rstd = rsqrtf(var + 1e-5f);
    const float w = sel[(size_t)t * N_EXP + e];
    #pragma unroll
    for (int j = 0; j < 2; ++j) {
        const int i = tid + j * 256;
        const float v = (r[j] - mean) * rstd * fng[i] + fnb[i];
        out[(size_t)t * D + i] += w * v;
    }
}

// ---------------------------------------------------------------------------
// Host launcher
// ---------------------------------------------------------------------------
static inline void launch_gemm(const float* A, const float* B, const float* bias,
                               float* C, int M, int N, int K, int epi,
                               hipStream_t stream) {
    dim3 grid(N / 128, M / 128), block(256);
    if (epi == 1)
        hipLaunchKernelGGL((gemm_wmma<1>), grid, block, 0, stream, A, B, bias, C, M, N, K);
    else
        hipLaunchKernelGGL((gemm_wmma<0>), grid, block, 0, stream, A, B, bias, C, M, N, K);
}

extern "C" void kernel_launch(void* const* d_in, const int* in_sizes, int n_in,
                              void* d_out, int out_size, void* d_ws, size_t ws_size,
                              hipStream_t stream) {
    (void)in_sizes; (void)n_in; (void)out_size; (void)ws_size;
    const float* x    = (const float*)d_in[0];
    const float* pw1  = (const float*)d_in[1];
    const float* pb1  = (const float*)d_in[2];
    const float* pg1  = (const float*)d_in[3];
    const float* pbb1 = (const float*)d_in[4];
    const float* pw2  = (const float*)d_in[5];
    const float* pb2  = (const float*)d_in[6];
    const float* Win  = (const float*)d_in[7];
    const float* b_in = (const float*)d_in[8];
    const float* W1   = (const float*)d_in[9];
    const float* b1   = (const float*)d_in[10];
    const float* W2   = (const float*)d_in[11];
    const float* b2   = (const float*)d_in[12];
    const float* ng   = (const float*)d_in[13];
    const float* nb   = (const float*)d_in[14];
    const float* gw   = (const float*)d_in[15];
    const float* Wout = (const float*)d_in[16];
    const float* bout = (const float*)d_in[17];
    const float* fng  = (const float*)d_in[18];
    const float* fnb  = (const float*)d_in[19];
    float* out = (float*)d_out;

    // workspace layout
    float* ws  = (float*)d_ws;
    float* Hp  = ws;                         // 2048*256
    float* sel = Hp  + (size_t)NTOK * 256;   // 2048*8
    float* H   = sel + (size_t)NTOK * N_EXP; // 2048*1024
    float* T   = H   + (size_t)NTOK * D_FF;  // 2048*2048
    float* U   = T   + (size_t)NTOK * 2 * D_FF; // 2048*1024
    float* O   = U   + (size_t)NTOK * D_FF;  // 2048*512

    // zero output accumulator
    hipLaunchKernelGGL(zero_kernel, dim3((NTOK * D_MODEL + 255) / 256), dim3(256),
                       0, stream, out, NTOK * D_MODEL);

    // --- policy head ---
    launch_gemm(x, pw1, pb1, Hp, NTOK, 256, D_MODEL, 0, stream);
    hipLaunchKernelGGL(policy_ln_gelu, dim3(NTOK), dim3(256), 0, stream, Hp, pg1, pbb1);
    hipLaunchKernelGGL(policy_route, dim3(NTOK), dim3(256), 0, stream, Hp, pw2, pb2, sel);

    // --- experts ---
    for (int e = 0; e < N_EXP; ++e) {
        launch_gemm(x, Win + (size_t)e * D_MODEL * D_FF, b_in + (size_t)e * D_FF,
                    H, NTOK, D_FF, D_MODEL, 0, stream);
        for (int l = 0; l < N_LAYERS; ++l) {
            const size_t el = (size_t)e * N_LAYERS + l;
            launch_gemm(H, W1 + el * (size_t)D_FF * 2 * D_FF, b1 + el * 2 * D_FF,
                        T, NTOK, 2 * D_FF, D_FF, /*gelu*/1, stream);
            launch_gemm(T, W2 + el * (size_t)2 * D_FF * D_FF, b2 + el * D_FF,
                        U, NTOK, D_FF, 2 * D_FF, 0, stream);
            hipLaunchKernelGGL(gate_ln_residual, dim3(NTOK), dim3(256), 0, stream,
                               H, U, ng + el * D_FF, nb + el * D_FF, gw + el * D_FF);
        }
        launch_gemm(H, Wout + (size_t)e * D_FF * D_MODEL, bout + (size_t)e * D_MODEL,
                    O, NTOK, D_MODEL, D_FF, 0, stream);
        hipLaunchKernelGGL(final_ln_accum, dim3(NTOK), dim3(256), 0, stream,
                           x, O, fng + (size_t)e * D_MODEL, fnb + (size_t)e * D_MODEL,
                           sel, e, out);
    }
}
